// TransformerEncoder_35313221108238
// MI455X (gfx1250) — compile-verified, matrix operand order checked by reading, not run on hardware
//
#include <hip/hip_runtime.h>
#include <hip/hip_bf16.h>
#include <math.h>

// ---------------- problem constants ----------------
#define EMB   1024
#define HEADS 16
#define DEPTH 4
#define BATCH 32
#define TSEQ  512
#define TOK   (BATCH * TSEQ)   // 16384 token rows
#define HDIM  (EMB / HEADS)    // 64
#define F_IN  512
#define CAT_K (3 * F_IN)       // 1536

typedef __bf16 bf16;
typedef __attribute__((ext_vector_type(16))) __bf16 v16bf;
typedef __attribute__((ext_vector_type(8)))  float  v8f;
typedef int v4i __attribute__((vector_size(16)));

#if __has_builtin(__builtin_amdgcn_global_load_async_to_lds_b128) && \
    __has_builtin(__builtin_amdgcn_s_wait_asynccnt)
#define HAVE_ASYNC_LDS 1
typedef __attribute__((address_space(1))) v4i* gptr_v4i;
typedef __attribute__((address_space(3))) v4i* lptr_v4i;
#else
#define HAVE_ASYNC_LDS 0
#endif

__device__ __forceinline__ bf16 f2bf(float x) {
  unsigned u = __builtin_bit_cast(unsigned, x);
  u += 0x7FFFu + ((u >> 16) & 1u);          // round-to-nearest-even
  unsigned short s = (unsigned short)(u >> 16);
  return __builtin_bit_cast(bf16, s);
}

// ---------------- elementwise helpers ----------------
__global__ void cvt_f32_bf16(const float* __restrict__ in, bf16* __restrict__ out, int n) {
  int i = blockIdx.x * blockDim.x + threadIdx.x;
  if (i < n) out[i] = f2bf(in[i]);
}

// concat [image1|image2|text] along feature axis -> bf16 A matrix [TOK, 1536]
__global__ void concat_kernel(const float* __restrict__ i1, const float* __restrict__ i2,
                              const float* __restrict__ tf, bf16* __restrict__ out) {
  int idx = blockIdx.x * blockDim.x + threadIdx.x;   // TOK*1536 total
  int c = idx % CAT_K;
  int row = idx / CAT_K;
  float v;
  if (c < F_IN)            v = i1[(size_t)row * F_IN + c];
  else if (c < 2 * F_IN)   v = i2[(size_t)row * F_IN + (c - F_IN)];
  else                     v = tf[(size_t)row * F_IN + (c - 2 * F_IN)];
  out[idx] = f2bf(v);
}

// ---------------- layernorm (one block per row) ----------------
__global__ __launch_bounds__(256) void layernorm_kernel(
    const float* __restrict__ X, const float* __restrict__ scale, const float* __restrict__ bias,
    bf16* __restrict__ outb, float* __restrict__ outf) {
  __shared__ float red[256];
  int row = blockIdx.x;
  const float* x = X + (size_t)row * EMB;
  float s = 0.f, s2 = 0.f;
  for (int c = threadIdx.x; c < EMB; c += 256) { float v = x[c]; s += v; s2 += v * v; }
  red[threadIdx.x] = s; __syncthreads();
  for (int st = 128; st > 0; st >>= 1) { if (threadIdx.x < st) red[threadIdx.x] += red[threadIdx.x + st]; __syncthreads(); }
  float mean = red[0] * (1.0f / EMB);
  __syncthreads();
  red[threadIdx.x] = s2; __syncthreads();
  for (int st = 128; st > 0; st >>= 1) { if (threadIdx.x < st) red[threadIdx.x] += red[threadIdx.x + st]; __syncthreads(); }
  float var = red[0] * (1.0f / EMB) - mean * mean;
  float rstd = rsqrtf(var + 1e-6f);
  for (int c = threadIdx.x; c < EMB; c += 256) {
    float v = (x[c] - mean) * rstd * scale[c] + bias[c];
    if (outb) outb[(size_t)row * EMB + c] = f2bf(v);
    if (outf) outf[(size_t)row * EMB + c] = v;
  }
}

// ---------------- bf16 WMMA GEMM with fused epilogue ----------------
// C[M,N] = A[M,K](bf16) * B[K,N](bf16), f32 accumulate via v_wmma_f32_16x16x32_bf16.
// Block: 256 threads = 8 wave32 waves (2 M x 4 N); block tile 128x256; wave tile 64x64
// (16 WMMAs per K-step of 32). A tile staged memory->LDS with CDNA5 async loads when
// available (ASYNCcnt), B tile staged transposed so K is LDS-contiguous for B-fragments.
#define BM 128
#define BN 256
#define BK 32
#define LDP 40   // LDS row pitch in bf16: 80B rows (16B aligned, 20 banks -> conflict-free b128)

enum { EPI_GELU = 1, EPI_POSEMB = 2 };

__global__ __launch_bounds__(256) void gemm_bf16_wmma(
    const bf16* __restrict__ A, const bf16* __restrict__ B,
    float* __restrict__ Cf, bf16* __restrict__ Cb,
    const float* __restrict__ bias, const float* __restrict__ resid,
    int M, int N, int K, int flags) {
  __shared__ bf16 As[BM][LDP];   // A tile, row-major
  __shared__ bf16 Bs[BN][LDP];   // B tile, transposed (N x K) so K is contiguous

  const int tid  = threadIdx.x;
  const int lane = tid & 31, wid = tid >> 5;
  const int half = lane >> 4, l15 = lane & 15, klo = half * 8;
  const int wm = wid >> 2, wn = wid & 3;
  const int m0 = blockIdx.y * BM, n0 = blockIdx.x * BN;

  v8f c[4][4];
  for (int mi = 0; mi < 4; ++mi)
    for (int ni = 0; ni < 4; ++ni)
      for (int e = 0; e < 8; ++e) c[mi][ni][e] = 0.f;

  for (int k0 = 0; k0 < K; k0 += BK) {
    __syncthreads();
    // stage A: 128x32 bf16, one 16B chunk per thread per pass, coalesced.
    // CDNA5 path: async memory->LDS copies tracked with ASYNCcnt.
    #pragma unroll
    for (int ch = 0; ch < 2; ++ch) {
      int q = ch * 256 + tid;               // 0..511
      int ar = q >> 2, ac = (q & 3) * 8;
      const bf16* src = A + (size_t)(m0 + ar) * K + k0 + ac;
#if HAVE_ASYNC_LDS
      __builtin_amdgcn_global_load_async_to_lds_b128(
          (gptr_v4i)src, (lptr_v4i)&As[ar][ac], 0, 0);
#else
      #pragma unroll
      for (int i = 0; i < 8; ++i) As[ar][ac + i] = src[i];
#endif
    }
    // stage B transposed: coalesced 16B reads along N, scatter into Bs[n][k]
    #pragma unroll
    for (int ch = 0; ch < 4; ++ch) {
      int q = ch * 256 + tid;               // 0..1023
      int nc = (q & 31) * 8, kr = q >> 5;   // nc: 0..248, kr: 0..31
      const bf16* src = B + (size_t)(k0 + kr) * N + n0 + nc;
      if (k0 + BK < K) __builtin_prefetch(src + (size_t)BK * N, 0, 0);
      #pragma unroll
      for (int i = 0; i < 8; ++i) Bs[nc + i][kr] = src[i];
    }
#if HAVE_ASYNC_LDS
    __builtin_amdgcn_s_wait_asynccnt(0);
#endif
    __syncthreads();

    // fragment loads (A layout: lane<16 holds K 0-7,16-23; lane>=16 holds K 8-15,24-31)
    v16bf a[4];
    #pragma unroll
    for (int mi = 0; mi < 4; ++mi) {
      const bf16* p = &As[wm * 64 + mi * 16 + l15][0];
      #pragma unroll
      for (int i = 0; i < 8; ++i) { a[mi][i] = p[klo + i]; a[mi][8 + i] = p[16 + klo + i]; }
    }
    #pragma unroll
    for (int ni = 0; ni < 4; ++ni) {
      v16bf b;
      const bf16* p = &Bs[wn * 64 + ni * 16 + l15][0];
      #pragma unroll
      for (int i = 0; i < 8; ++i) { b[i] = p[klo + i]; b[8 + i] = p[16 + klo + i]; }
      #pragma unroll
      for (int mi = 0; mi < 4; ++mi)
        c[mi][ni] = __builtin_amdgcn_wmma_f32_16x16x32_bf16(
            false, a[mi], false, b, (short)0, c[mi][ni], false, false);
    }
  }

  // fused epilogue: bias / sincos pos-embed / GELU / residual, store f32 and/or bf16
  #pragma unroll
  for (int mi = 0; mi < 4; ++mi) {
    #pragma unroll
    for (int ni = 0; ni < 4; ++ni) {
      int gcol = n0 + wn * 64 + ni * 16 + l15;
      #pragma unroll
      for (int r = 0; r < 8; ++r) {
        int grow = m0 + wm * 64 + mi * 16 + r + half * 8;
        float v = c[mi][ni][r];
        if (bias) v += bias[gcol];
        if (flags & EPI_POSEMB) {
          int t = grow & (TSEQ - 1);
          int j = gcol & (EMB / 2 - 1);
          float omega = __expf(-(float)j * (9.210340371976184f / (EMB / 2)));
          float ang = (float)t * omega;
          v += (gcol < EMB / 2) ? __sinf(ang) : __cosf(ang);
        }
        if (flags & EPI_GELU) {
          float x = v;
          v = 0.5f * x * (1.0f + tanhf(0.7978845608f * (x + 0.044715f * x * x * x)));
        }
        if (resid) v += resid[(size_t)grow * N + gcol];
        if (Cf) Cf[(size_t)grow * N + gcol] = v;
        if (Cb) Cb[(size_t)grow * N + gcol] = f2bf(v);
      }
    }
  }
}

// ---------------- fused flash attention ----------------
// grid (T/64, B*H); block 128 = 4 waves, each wave owns 16 q-rows.
// QKV buffer is bf16 [TOK, 3*EMB]; output attention result bf16 [TOK, EMB].
__global__ __launch_bounds__(128) void attn_kernel(const bf16* __restrict__ qkv,
                                                   bf16* __restrict__ out) {
  __shared__ bf16 VT[HDIM][LDP];       // V^T tile: 64 dims x 32 keys
  __shared__ bf16 Plds[4][16][LDP];    // per-wave P tile 16 q-rows x 32 keys

  const int tid  = threadIdx.x;
  const int lane = tid & 31, w = tid >> 5;
  const int half = lane >> 4, l15 = lane & 15, klo = half * 8;
  const int bh = blockIdx.y, b = bh >> 4, h = bh & 15;
  const int q0 = blockIdx.x * 64;
  const size_t rs = 3 * EMB;
  const int qcol = h * HDIM, kcol = EMB + h * HDIM, vcol = 2 * EMB + h * HDIM;
  const size_t base = (size_t)b * TSEQ * rs;

  // preload Q A-fragments for this wave's 16 rows (head_dim 64 -> 2 K-steps)
  v16bf qa[2];
  {
    const bf16* qrow = qkv + base + (size_t)(q0 + w * 16 + l15) * rs + qcol;
    #pragma unroll
    for (int s = 0; s < 2; ++s) {
      const bf16* p = qrow + s * 32;
      #pragma unroll
      for (int i = 0; i < 8; ++i) { qa[s][i] = p[klo + i]; qa[s][8 + i] = p[16 + klo + i]; }
    }
  }

  v8f o[4];
  for (int n = 0; n < 4; ++n) for (int e = 0; e < 8; ++e) o[n][e] = 0.f;
  float mrow[8], lrow[8];
  for (int r = 0; r < 8; ++r) { mrow[r] = -1e30f; lrow[r] = 0.f; }

  for (int kb = 0; kb < TSEQ; kb += 32) {
    __syncthreads();
    // stage V^T (32 keys x 64 dims -> VT[d][j]); contiguous 16B reads per chunk
    #pragma unroll
    for (int ch = 0; ch < 2; ++ch) {
      int q = ch * 128 + tid;              // 0..255
      int d8 = (q & 7) * 8, j = q >> 3;    // d8: 0..56, j: 0..31
      const bf16* src = qkv + base + (size_t)(kb + j) * rs + vcol + d8;
      #pragma unroll
      for (int i = 0; i < 8; ++i) VT[d8 + i][j] = src[i];
    }
    __syncthreads();

    // S = Q * K^T (16 x 32); K rows are contiguous -> direct global fragment loads
    v8f s[2];
    for (int ni = 0; ni < 2; ++ni) for (int e = 0; e < 8; ++e) s[ni][e] = 0.f;
    #pragma unroll
    for (int ks = 0; ks < 2; ++ks) {
      #pragma unroll
      for (int ni = 0; ni < 2; ++ni) {
        v16bf kf;
        const bf16* p = qkv + base + (size_t)(kb + ni * 16 + l15) * rs + kcol + ks * 32;
        #pragma unroll
        for (int i = 0; i < 8; ++i) { kf[i] = p[klo + i]; kf[8 + i] = p[16 + klo + i]; }
        s[ni] = __builtin_amdgcn_wmma_f32_16x16x32_bf16(
            false, qa[ks], false, kf, (short)0, s[ni], false, false);
      }
    }
    #pragma unroll
    for (int ni = 0; ni < 2; ++ni)
      #pragma unroll
      for (int r = 0; r < 8; ++r) s[ni][r] *= 0.125f;   // 1/sqrt(64)

    // online softmax: row r lives in lanes 0-15 (and row r+8 in lanes 16-31)
    #pragma unroll
    for (int r = 0; r < 8; ++r) {
      float mx = fmaxf(s[0][r], s[1][r]);
      #pragma unroll
      for (int msk = 8; msk; msk >>= 1) mx = fmaxf(mx, __shfl_xor(mx, msk, 16));
      float mnew = fmaxf(mrow[r], mx);
      float alpha = __expf(mrow[r] - mnew);
      mrow[r] = mnew;
      float psum = 0.f;
      #pragma unroll
      for (int ni = 0; ni < 2; ++ni) {
        float p = __expf(s[ni][r] - mnew);
        psum += p;
        Plds[w][r + half * 8][ni * 16 + l15] = f2bf(p);  // C-layout -> LDS
      }
      #pragma unroll
      for (int msk = 8; msk; msk >>= 1) psum += __shfl_xor(psum, msk, 16);
      lrow[r] = lrow[r] * alpha + psum;
      #pragma unroll
      for (int n = 0; n < 4; ++n) o[n][r] *= alpha;
    }

    // reload P as an A-fragment from LDS (per-wave region, wave-ordered LDS)
    v16bf pa;
    {
      const bf16* p = &Plds[w][l15][0];
      #pragma unroll
      for (int i = 0; i < 8; ++i) { pa[i] = p[klo + i]; pa[8 + i] = p[16 + klo + i]; }
    }
    // O += P(16x32) * V(32x64): B-fragments from transposed V tile
    #pragma unroll
    for (int n = 0; n < 4; ++n) {
      v16bf vf;
      const bf16* p = &VT[n * 16 + l15][0];
      #pragma unroll
      for (int i = 0; i < 8; ++i) { vf[i] = p[klo + i]; vf[8 + i] = p[16 + klo + i]; }
      o[n] = __builtin_amdgcn_wmma_f32_16x16x32_bf16(
          false, pa, false, vf, (short)0, o[n], false, false);
    }
  }

  // normalize and store bf16 (feeds proj GEMM directly)
  #pragma unroll
  for (int n = 0; n < 4; ++n) {
    int gcol = h * HDIM + n * 16 + l15;
    #pragma unroll
    for (int r = 0; r < 8; ++r) {
      int grow = b * TSEQ + q0 + w * 16 + r + half * 8;
      out[(size_t)grow * EMB + gcol] = f2bf(o[n][r] / lrow[r]);
    }
  }
}

// ---------------- host orchestration ----------------
extern "C" void kernel_launch(void* const* d_in, const int* in_sizes, int n_in,
                              void* d_out, int out_size, void* d_ws, size_t ws_size,
                              hipStream_t stream) {
  const float* image1     = (const float*)d_in[0];
  const float* image2     = (const float*)d_in[1];
  const float* text       = (const float*)d_in[2];
  const float* mlp_kernel = (const float*)d_in[3];
  const float* mlp_bias   = (const float*)d_in[4];
  const float* ln1_scale  = (const float*)d_in[5];
  const float* ln1_bias   = (const float*)d_in[6];
  const float* qkv_kernel = (const float*)d_in[7];
  const float* qkv_bias   = (const float*)d_in[8];
  const float* proj_kernel= (const float*)d_in[9];
  const float* proj_bias  = (const float*)d_in[10];
  const float* ln2_scale  = (const float*)d_in[11];
  const float* ln2_bias   = (const float*)d_in[12];
  const float* fc1_kernel = (const float*)d_in[13];
  const float* fc2_kernel = (const float*)d_in[14];
  const float* lnf_scale  = (const float*)d_in[15];
  const float* lnf_bias   = (const float*)d_in[16];

  char* ws = (char*)d_ws;
  size_t off = 0;
  auto carve = [&](size_t bytes) -> void* {
    void* p = ws + off;
    off += (bytes + 255) & ~(size_t)255;
    return p;
  };
  float* X   = (float*)carve((size_t)TOK * EMB * 4);         // f32 residual stream (64 MB)
  bf16*  A0  = (bf16*) carve((size_t)TOK * CAT_K * 2);       // bf16 activations (48 MB)
  bf16*  A1  = (bf16*) carve((size_t)TOK * 4 * EMB * 2);     // fc1 output (128 MB)
  bf16*  QKV = (bf16*) carve((size_t)TOK * 3 * EMB * 2);     // qkv bf16 (96 MB)
  bf16*  Wb  = (bf16*) carve((size_t)4 * EMB * EMB * 2);     // weight scratch (8 MB)

  auto cvt = [&](const float* src, bf16* dst, int n) {
    cvt_f32_bf16<<<(n + 255) / 256, 256, 0, stream>>>(src, dst, n);
  };
  auto gemm = [&](const bf16* A, const bf16* Bw, float* Cf, bf16* Cb,
                  const float* bias, const float* resid, int M, int N, int K, int flags) {
    dim3 g(N / BN, M / BM);
    gemm_bf16_wmma<<<g, 256, 0, stream>>>(A, Bw, Cf, Cb, bias, resid, M, N, K, flags);
  };

  // 1) feature concat -> bf16, input MLP + bias + sincos pos-embed -> X (f32)
  concat_kernel<<<(TOK * CAT_K) / 256, 256, 0, stream>>>(image1, image2, text, A0);
  cvt(mlp_kernel, Wb, CAT_K * EMB);
  gemm(A0, Wb, X, nullptr, mlp_bias, nullptr, TOK, EMB, CAT_K, EPI_POSEMB);

  for (int i = 0; i < DEPTH; ++i) {
    // attention block
    layernorm_kernel<<<TOK, 256, 0, stream>>>(X, ln1_scale + i * EMB, ln1_bias + i * EMB, A0, nullptr);
    cvt(qkv_kernel + (size_t)i * EMB * 3 * EMB, Wb, EMB * 3 * EMB);
    gemm(A0, Wb, nullptr, QKV, qkv_bias + i * 3 * EMB, nullptr, TOK, 3 * EMB, EMB, 0);
    attn_kernel<<<dim3(TSEQ / 64, BATCH * HEADS), 128, 0, stream>>>(QKV, A0);
    cvt(proj_kernel + (size_t)i * EMB * EMB, Wb, EMB * EMB);
    gemm(A0, Wb, X, nullptr, proj_bias + i * EMB, X, TOK, EMB, EMB, 0);  // + residual
    // MLP block
    layernorm_kernel<<<TOK, 256, 0, stream>>>(X, ln2_scale + i * EMB, ln2_bias + i * EMB, A0, nullptr);
    cvt(fc1_kernel + (size_t)i * EMB * 4 * EMB, Wb, EMB * 4 * EMB);
    gemm(A0, Wb, nullptr, A1, nullptr, nullptr, TOK, 4 * EMB, EMB, EPI_GELU);
    cvt(fc2_kernel + (size_t)i * 4 * EMB * EMB, Wb, 4 * EMB * EMB);
    gemm(A1, Wb, X, nullptr, nullptr, X, TOK, EMB, 4 * EMB, 0);          // + residual
  }
  // final layernorm -> f32 output
  layernorm_kernel<<<TOK, 256, 0, stream>>>(X, lnf_scale, lnf_bias, nullptr, (float*)d_out);
}
